// AdaAttN_15573551415829
// MI455X (gfx1250) — matrix operand order repaired and checked by reading, not compiled, not run on hardware
//
#include <hip/hip_runtime.h>
#include <math.h>

// Problem constants (from reference setup_inputs): b=4, C=qk=v=64, N=64*64=4096
#define BB 4
#define CD 64
#define NP 4096
#define EPS 1e-5f

typedef _Float16 half_t;
typedef __attribute__((ext_vector_type(16))) _Float16 v16h;
typedef __attribute__((ext_vector_type(8)))  _Float16 v8h;
typedef __attribute__((ext_vector_type(8)))  float    v8f;

__device__ inline v16h cat8(v8h lo, v8h hi) {
  return __builtin_shufflevector(lo, hi, 0,1,2,3,4,5,6,7,8,9,10,11,12,13,14,15);
}

// ---------------------------------------------------------------------------
// Kernel 1: instance-norm statistics (mean, rstd) per (b, ch) row of q, k, c.
// grid = 3*256 blocks (tsel*256 + b*64 + c), 256 threads reduce 4096 elements.
// stats layout: stats[(tsel*256 + b*64 + c)*2 + {0:mean,1:rstd}]
// ---------------------------------------------------------------------------
__global__ __launch_bounds__(256) void stats_kernel(
    const float* __restrict__ q, const float* __restrict__ k,
    const float* __restrict__ c, float* __restrict__ stats)
{
  const int id   = blockIdx.x;
  const int tsel = id >> 8;
  const int row  = id & 255;
  const float* x = (tsel == 0 ? q : tsel == 1 ? k : c) + (size_t)row * NP;

  float s = 0.f, s2 = 0.f;
  for (int i = threadIdx.x; i < NP; i += 256) {
    float v = x[i]; s += v; s2 += v * v;
  }
  __shared__ float sh0[256], sh1[256];
  sh0[threadIdx.x] = s; sh1[threadIdx.x] = s2;
  __syncthreads();
  for (int off = 128; off > 0; off >>= 1) {
    if ((int)threadIdx.x < off) {
      sh0[threadIdx.x] += sh0[threadIdx.x + off];
      sh1[threadIdx.x] += sh1[threadIdx.x + off];
    }
    __syncthreads();
  }
  if (threadIdx.x == 0) {
    float m   = sh0[0] * (1.0f / NP);
    float var = sh1[0] * (1.0f / NP) - m * m;
    stats[id * 2 + 0] = m;
    stats[id * 2 + 1] = rsqrtf(var + EPS);
  }
}

// ---------------------------------------------------------------------------
// Kernel 2: fused (optional instance-norm) + 1x1 conv, output f16.
//   transposeOut=1: out[(b*NP+n)*64 + co]   (for q,k -> WMMA A/B over channels)
//   transposeOut=0: out[(b*64+co)*NP + n]   (for s   -> WMMA B over keys)
//                   plus outSq = out^2 in same layout.
// grid = B*(NP/64) = 256 blocks, 256 threads; block handles 64 positions.
// ---------------------------------------------------------------------------
__global__ __launch_bounds__(256) void proj_kernel(
    const float* __restrict__ X,     // [B][64][NP]
    const float* __restrict__ W,     // [64][64]
    const float* __restrict__ bias,  // [64]
    const float* __restrict__ stats, // per-row (mean,rstd), or nullptr
    half_t* __restrict__ outA,
    half_t* __restrict__ outSq,      // may be nullptr
    int transposeOut)
{
  __shared__ float Wsh[64][64];
  __shared__ float Xsh[64][65];
  const int b  = blockIdx.x >> 6;
  const int n0 = (blockIdx.x & 63) << 6;
  const int t  = threadIdx.x;

  #pragma unroll
  for (int i = 0; i < 16; ++i) {
    int idx = t + 256 * i;                 // 0..4095
    Wsh[idx >> 6][idx & 63] = W[idx];
  }
  for (int i = 0; i < 16; ++i) {
    int idx = t + 256 * i;
    int ci = idx >> 6, p = idx & 63;
    float v = X[((size_t)(b * 64 + ci)) * NP + n0 + p];
    if (stats) {
      float m = stats[(b * 64 + ci) * 2 + 0];
      float r = stats[(b * 64 + ci) * 2 + 1];
      v = (v - m) * r;
    }
    Xsh[ci][p] = v;
  }
  __syncthreads();

  const int co = t >> 2;                   // 0..63
  const int p4 = t & 3;
  #pragma unroll
  for (int i = 0; i < 16; ++i) {
    int p = p4 + 4 * i;
    float acc = bias[co];
    #pragma unroll 8
    for (int ci = 0; ci < 64; ++ci) acc += Wsh[co][ci] * Xsh[ci][p];
    if (transposeOut) {
      outA[((size_t)(b * NP + n0 + p)) * 64 + co] = (half_t)acc;
    } else {
      size_t idx = ((size_t)(b * 64 + co)) * NP + n0 + p;
      outA[idx] = (half_t)acc;
      if (outSq) outSq[idx] = (half_t)(acc * acc);
    }
  }
}

// ---------------------------------------------------------------------------
// Kernel 3: flash attention producing per-query mean & std of attended style.
// grid = B*(NP/64) = 256 blocks, 128 threads = 4 waves; wave owns 16 q-rows.
// Key block = 64: 8 WMMA (scores, K=64) + 16 WMMA (P*s, P*s^2) per iteration.
// Softmax denominators kept as per-lane partials; lane-reduced once at end.
// ---------------------------------------------------------------------------
__global__ __launch_bounds__(128) void flash_attn_kernel(
    const half_t* __restrict__ qeT,  // [B][NP][64]
    const half_t* __restrict__ keT,  // [B][NP][64]
    const half_t* __restrict__ seB,  // [B][64][NP]
    const half_t* __restrict__ s2B,  // [B][64][NP]
    float* __restrict__ meanBuf,     // [B][NP][64]
    float* __restrict__ stdBuf)      // [B][NP][64]
{
  __shared__ __align__(64) half_t Pbuf[4][16][64];
  const int wave  = threadIdx.x >> 5;
  const int lane  = threadIdx.x & 31;
  const int ln16  = lane & 15;
  const int hsel  = lane >> 4;                 // half-wave select
  const int b     = blockIdx.x >> 6;
  const int qBase = ((blockIdx.x & 63) << 6) + (wave << 4);

  // Preload q A-fragments (16x32 f16 layout): lane holds row qBase+ln16,
  // K chunk [16*hsel,+16) within each 32-wide c-chunk.
  const half_t* qrow = qeT + ((size_t)(b * NP + qBase + ln16)) * 64;
  v16h qA[2];
  #pragma unroll
  for (int kk = 0; kk < 2; ++kk) {
    const half_t* p = qrow + 32 * kk;
    qA[kk] = *(const v16h*)(p + 16 * hsel);
  }

  float rm[8], rs[8];                // rs = per-lane PARTIAL sums (16 cols)
  v8f Om[4] = {}, Oq[4] = {};
  #pragma unroll
  for (int r = 0; r < 8; ++r) { rm[r] = -3.0e38f; rs[r] = 0.f; }

  for (int kb = 0; kb < NP / 64; ++kb) {
    const int keyBase = kb * 64;

    // ---- scores: four 16x16 f32 tiles over 64 keys, K = 64 channels ----
    v8f S[4];
    #pragma unroll
    for (int tt = 0; tt < 4; ++tt) {
      const half_t* krow = keT + ((size_t)(b * NP + keyBase + 16 * tt + ln16)) * 64;
      v16h B0 = *(const v16h*)(krow + 16 * hsel);        // c = 0..31 chunk
      v16h B1 = *(const v16h*)(krow + 32 + 16 * hsel);   // c = 32..63 chunk
      v8f acc = {};
      acc = __builtin_amdgcn_wmma_f32_16x16x32_f16(false, qA[0], false, B0,
                                                   (short)0, acc, false, false);
      acc = __builtin_amdgcn_wmma_f32_16x16x32_f16(false, qA[1], false, B1,
                                                   (short)0, acc, false, false);
      S[tt] = acc;
    }

    // ---- online softmax: only the MAX needs a lane reduction ----
    #pragma unroll
    for (int r = 0; r < 8; ++r) {
      float v = fmaxf(fmaxf(S[0][r], S[1][r]), fmaxf(S[2][r], S[3][r]));
      v = fmaxf(v, __shfl_xor(v, 1));
      v = fmaxf(v, __shfl_xor(v, 2));
      v = fmaxf(v, __shfl_xor(v, 4));
      v = fmaxf(v, __shfl_xor(v, 8));
      float nm    = fmaxf(rm[r], v);
      float scale = __expf(rm[r] - nm);
      rm[r] = nm;
      float psum = 0.f;
      #pragma unroll
      for (int tt = 0; tt < 4; ++tt) {
        float p = __expf(S[tt][r] - nm);
        S[tt][r] = p;
        psum += p;
      }
      rs[r] = rs[r] * scale + psum;        // lane-local partial denominator
      #pragma unroll
      for (int g = 0; g < 4; ++g) {
        Om[g][r] = Om[g][r] * scale;
        Oq[g][r] = Oq[g][r] * scale;
      }
    }

    // ---- C-layout P -> A-layout via LDS ----
    __syncthreads();                       // WAR vs previous read
    #pragma unroll
    for (int r = 0; r < 8; ++r) {
      int row = r + 8 * hsel;
      Pbuf[wave][row][ln16]      = (half_t)S[0][r];
      Pbuf[wave][row][16 + ln16] = (half_t)S[1][r];
      Pbuf[wave][row][32 + ln16] = (half_t)S[2][r];
      Pbuf[wave][row][48 + ln16] = (half_t)S[3][r];
    }
    __syncthreads();

    // ---- accumulate mean and E[s^2]: two 32-key P fragments x 4 groups ----
    #pragma unroll
    for (int kc = 0; kc < 2; ++kc) {
      v16h pA;
      {
        const half_t* pr = &Pbuf[wave][ln16][32 * kc];
        pA = *(const v16h*)(pr + 16 * hsel);
      }
      #pragma unroll
      for (int g = 0; g < 4; ++g) {
        size_t base = ((size_t)(b * 64 + 16 * g + ln16)) * NP
                    + keyBase + 32 * kc + 16 * hsel;
        v16h Bs  = *(const v16h*)(seB + base);
        v16h Bs2 = *(const v16h*)(s2B + base);
        Om[g] = __builtin_amdgcn_wmma_f32_16x16x32_f16(false, pA, false, Bs,
                                                       (short)0, Om[g], false, false);
        Oq[g] = __builtin_amdgcn_wmma_f32_16x16x32_f16(false, pA, false, Bs2,
                                                       (short)0, Oq[g], false, false);
      }
    }
  }

  // ---- final lane reduction of the deferred denominators ----
  #pragma unroll
  for (int r = 0; r < 8; ++r) {
    float t = rs[r];
    t += __shfl_xor(t, 1);
    t += __shfl_xor(t, 2);
    t += __shfl_xor(t, 4);
    t += __shfl_xor(t, 8);
    rs[r] = t;
  }

  // ---- finalize: mean, std = sqrt(relu(E[s^2] - mean^2)) ----
  #pragma unroll
  for (int r = 0; r < 8; ++r) {
    float inv = 1.0f / rs[r];
    int row = qBase + r + 8 * hsel;
    #pragma unroll
    for (int g = 0; g < 4; ++g) {
      float mv  = Om[g][r] * inv;
      float msq = Oq[g][r] * inv;
      float var = fmaxf(msq - mv * mv, 0.f);
      size_t idx = ((size_t)(b * NP + row)) * 64 + 16 * g + ln16;
      meanBuf[idx] = mv;
      stdBuf[idx]  = sqrtf(var);
    }
  }
}

// ---------------------------------------------------------------------------
// Kernel 4: out[b][c][n] = inorm(c)[b][c][n] * std[b][n][c] + mean[b][n][c]
// ---------------------------------------------------------------------------
__global__ __launch_bounds__(256) void epilogue_kernel(
    const float* __restrict__ cIn,     // [B][64][NP]
    const float* __restrict__ stats,   // full stats buffer; c rows at +512
    const float* __restrict__ meanBuf, // [B][NP][64]
    const float* __restrict__ stdBuf,
    float* __restrict__ out)
{
  size_t idx = (size_t)blockIdx.x * 256 + threadIdx.x;  // over B*64*NP
  int n   = (int)(idx & (NP - 1));
  int bc  = (int)(idx >> 12);      // b*64 + c
  int cch = bc & 63;
  int b   = bc >> 6;
  float m = stats[(512 + bc) * 2 + 0];
  float r = stats[(512 + bc) * 2 + 1];
  float xn = (cIn[idx] - m) * r;
  size_t sidx = ((size_t)(b * NP + n)) * 64 + cch;
  out[idx] = xn * stdBuf[sidx] + meanBuf[sidx];
}

// ---------------------------------------------------------------------------
// Launch.  Workspace requirement: 4*(B*NP*64*2) + 2*(B*NP*64*4) + 768*2*4
//        = 8 MiB (f16 operands) + 8 MiB (mean/std f32) + 6 KiB  ~= 16.8 MiB
// ---------------------------------------------------------------------------
extern "C" void kernel_launch(void* const* d_in, const int* in_sizes, int n_in,
                              void* d_out, int out_size, void* d_ws, size_t ws_size,
                              hipStream_t stream) {
  const float* q   = (const float*)d_in[0];
  const float* k   = (const float*)d_in[1];
  const float* c   = (const float*)d_in[2];
  const float* s   = (const float*)d_in[3];
  const float* Wq  = (const float*)d_in[4];
  const float* bq  = (const float*)d_in[5];
  const float* Wk  = (const float*)d_in[6];
  const float* bk  = (const float*)d_in[7];
  const float* Ws  = (const float*)d_in[8];
  const float* bsv = (const float*)d_in[9];
  float* out = (float*)d_out;

  char* ws = (char*)d_ws;
  const size_t halfBuf = (size_t)BB * NP * 64 * sizeof(half_t);
  const size_t f32Buf  = (size_t)BB * NP * 64 * sizeof(float);
  half_t* qeT = (half_t*)ws;              ws += halfBuf;
  half_t* keT = (half_t*)ws;              ws += halfBuf;
  half_t* seB = (half_t*)ws;              ws += halfBuf;
  half_t* s2B = (half_t*)ws;              ws += halfBuf;
  float* meanBuf = (float*)ws;            ws += f32Buf;
  float* stdBuf  = (float*)ws;            ws += f32Buf;
  float* stats   = (float*)ws;            // 768*2 floats

  // 1) instance-norm stats for q (rows 0..255), k (256..511), c (512..767)
  stats_kernel<<<768, 256, 0, stream>>>(q, k, c, stats);

  // 2) projections to f16 WMMA layouts
  proj_kernel<<<BB * (NP / 64), 256, 0, stream>>>(q, Wq, bq, stats,        qeT, nullptr, 1);
  proj_kernel<<<BB * (NP / 64), 256, 0, stream>>>(k, Wk, bk, stats + 512,  keT, nullptr, 1);
  proj_kernel<<<BB * (NP / 64), 256, 0, stream>>>(s, Ws, bsv, nullptr,     seB, s2B,     0);

  // 3) fused flash attention -> per-position mean/std of attended style
  flash_attn_kernel<<<BB * (NP / 64), 128, 0, stream>>>(qeT, keT, seB, s2B,
                                                        meanBuf, stdBuf);

  // 4) cs = inorm(c) * std + mean
  epilogue_kernel<<<(BB * 64 * NP) / 256, 256, 0, stream>>>(c, stats, meanBuf,
                                                            stdBuf, out);
}
